// DatTransformerBlock_90357521973521
// MI455X (gfx1250) — compile-verified, matrix operand order checked by reading, not run on hardware
//
#include <hip/hip_runtime.h>

typedef __attribute__((ext_vector_type(16))) _Float16 v16h;
typedef __attribute__((ext_vector_type(4)))  _Float16 v4h;
typedef __attribute__((ext_vector_type(8)))  float    v8f;

#define B_    4
#define C_    256
#define H_    48
#define W_    48
#define HW_   (H_*W_)     // 2304
#define HEADS_ 8
#define HC_   32
#define G_    4
#define GC_   64
#define GH_   2
#define HK_   24
#define WK_   24
#define NS_   (HK_*WK_)   // 576
#define RT_   95          // 2*H-1
#define SCALE_ 0.17677669529663687f
#define EPS_   1e-5f

// ---------------------------------------------------------------------------
// 1x1-conv GEMM: Y[b][o][p] = sum_c X[b][c][p] * Wm[o*C_+c] + bias[o]
// fp32 in/out, f16 WMMA (f32 accumulate). Block = 128 thr = 4 waves.
// Block tile: M=64 pixels x N=64 out-channels, K=256 in 32-chunks.
// Staging: all 8 float4 global loads issued into distinct registers before
// any conversion/LDS store, so one wait drains them all. Fragment reads are
// ds_load_b128; 4 WMMAs per K-chunk per wave.
// ---------------------------------------------------------------------------
template<int P>
__global__ void __launch_bounds__(128) proj_kernel(const float* __restrict__ X,
                                                   const float* __restrict__ Wm,
                                                   const float* __restrict__ bias,
                                                   float* __restrict__ Y) {
  __shared__ _Float16 At[64][40];   // [m][k], k padded 32->40 (16B-aligned rows)
  __shared__ _Float16 Bt[64][40];   // [o][k]
  const int tid  = threadIdx.x;
  const int wave = tid >> 5, lane = tid & 31;
  const int col  = lane & 15;
  const int hlf  = lane >> 4;                     // 0/1
  const int kb   = hlf * 8;
  const int p0   = blockIdx.x * 64;
  const int o0   = blockIdx.y * 64;
  const int bb   = blockIdx.z;

  const float* Xb = X + (size_t)bb * C_ * P;
  v8f acc[4] = {};
  for (int k0 = 0; k0 < C_; k0 += 32) {
    // ---- load phase: 8 independent float4 loads, all in flight ----
    float4 xa[4], wa[4];
    #pragma unroll
    for (int i = 0; i < 4; ++i) {
      int gi = tid + i * 128;                     // 512 float4 groups
      int c = gi >> 4, mg = (gi & 15) * 4;
      xa[i] = *(const float4*)(Xb + (size_t)(k0 + c) * P + p0 + mg);
    }
    #pragma unroll
    for (int i = 0; i < 4; ++i) {
      int gi = tid + i * 128;
      int o = gi >> 3, cg = (gi & 7) * 4;
      wa[i] = *(const float4*)(Wm + (size_t)(o0 + o) * C_ + k0 + cg);
    }
    // ---- convert + store phase ----
    #pragma unroll
    for (int i = 0; i < 4; ++i) {
      int gi = tid + i * 128;
      int c = gi >> 4, mg = (gi & 15) * 4;
      At[mg + 0][c] = (_Float16)xa[i].x;
      At[mg + 1][c] = (_Float16)xa[i].y;
      At[mg + 2][c] = (_Float16)xa[i].z;
      At[mg + 3][c] = (_Float16)xa[i].w;
    }
    #pragma unroll
    for (int i = 0; i < 4; ++i) {
      int gi = tid + i * 128;
      int o = gi >> 3, cg = (gi & 7) * 4;
      v4h hv;
      hv[0] = (_Float16)wa[i].x; hv[1] = (_Float16)wa[i].y;
      hv[2] = (_Float16)wa[i].z; hv[3] = (_Float16)wa[i].w;
      *(v4h*)&Bt[o][cg] = hv;
    }
    __syncthreads();
    const int mrow = wave * 16 + col;
    v16h a;
    #pragma unroll
    for (int i = 0; i < 8; ++i) a[i]     = At[mrow][kb + i];
    #pragma unroll
    for (int i = 0; i < 8; ++i) a[8 + i] = At[mrow][kb + 16 + i];
    #pragma unroll
    for (int j = 0; j < 4; ++j) {
      const int orow = j * 16 + col;
      v16h bfr;
      #pragma unroll
      for (int i = 0; i < 8; ++i) bfr[i]     = Bt[orow][kb + i];
      #pragma unroll
      for (int i = 0; i < 8; ++i) bfr[8 + i] = Bt[orow][kb + 16 + i];
      acc[j] = __builtin_amdgcn_wmma_f32_16x16x32_f16(false, a, false, bfr,
                                                      (short)0, acc[j], false, false);
    }
    __syncthreads();
  }
  // D frags: row = v + 8*hlf (consecutive p), col = out channel (j*16+col)
  #pragma unroll
  for (int j = 0; j < 4; ++j) {
    const int o = o0 + j * 16 + col;
    const float bo = bias[o];
    float* yp = Y + ((size_t)bb * C_ + o) * P + p0 + wave * 16 + hlf * 8;
    #pragma unroll
    for (int v = 0; v < 8; ++v) yp[v] = acc[j][v] + bo;
  }
}

// ---------------------------------------------------------------------------
// k/v projection from x_s [B][C][NS] fp32 -> f16 outputs. Same M64xN64 tiling.
// transposed=1: kT[bh][n][c] ; transposed=0: vC[(b*C+o)][n]
// ---------------------------------------------------------------------------
__global__ void __launch_bounds__(128) kv_kernel(const float* __restrict__ X,
                                                 const float* __restrict__ Wm,
                                                 const float* __restrict__ bias,
                                                 _Float16* __restrict__ out,
                                                 int transposed) {
  __shared__ _Float16 At[64][40];
  __shared__ _Float16 Bt[64][40];
  const int tid  = threadIdx.x;
  const int wave = tid >> 5, lane = tid & 31;
  const int col  = lane & 15;
  const int hlf  = lane >> 4;
  const int kb   = hlf * 8;
  const int p0   = blockIdx.x * 64;
  const int o0   = blockIdx.y * 64;
  const int bb   = blockIdx.z;

  const float* Xb = X + (size_t)bb * C_ * NS_;
  v8f acc[4] = {};
  for (int k0 = 0; k0 < C_; k0 += 32) {
    float4 xa[4], wa[4];
    #pragma unroll
    for (int i = 0; i < 4; ++i) {
      int gi = tid + i * 128;
      int c = gi >> 4, mg = (gi & 15) * 4;
      xa[i] = *(const float4*)(Xb + (size_t)(k0 + c) * NS_ + p0 + mg);
    }
    #pragma unroll
    for (int i = 0; i < 4; ++i) {
      int gi = tid + i * 128;
      int o = gi >> 3, cg = (gi & 7) * 4;
      wa[i] = *(const float4*)(Wm + (size_t)(o0 + o) * C_ + k0 + cg);
    }
    #pragma unroll
    for (int i = 0; i < 4; ++i) {
      int gi = tid + i * 128;
      int c = gi >> 4, mg = (gi & 15) * 4;
      At[mg + 0][c] = (_Float16)xa[i].x;
      At[mg + 1][c] = (_Float16)xa[i].y;
      At[mg + 2][c] = (_Float16)xa[i].z;
      At[mg + 3][c] = (_Float16)xa[i].w;
    }
    #pragma unroll
    for (int i = 0; i < 4; ++i) {
      int gi = tid + i * 128;
      int o = gi >> 3, cg = (gi & 7) * 4;
      v4h hv;
      hv[0] = (_Float16)wa[i].x; hv[1] = (_Float16)wa[i].y;
      hv[2] = (_Float16)wa[i].z; hv[3] = (_Float16)wa[i].w;
      *(v4h*)&Bt[o][cg] = hv;
    }
    __syncthreads();
    const int mrow = wave * 16 + col;
    v16h a;
    #pragma unroll
    for (int i = 0; i < 8; ++i) a[i]     = At[mrow][kb + i];
    #pragma unroll
    for (int i = 0; i < 8; ++i) a[8 + i] = At[mrow][kb + 16 + i];
    #pragma unroll
    for (int j = 0; j < 4; ++j) {
      const int orow = j * 16 + col;
      v16h bfr;
      #pragma unroll
      for (int i = 0; i < 8; ++i) bfr[i]     = Bt[orow][kb + i];
      #pragma unroll
      for (int i = 0; i < 8; ++i) bfr[8 + i] = Bt[orow][kb + 16 + i];
      acc[j] = __builtin_amdgcn_wmma_f32_16x16x32_f16(false, a, false, bfr,
                                                      (short)0, acc[j], false, false);
    }
    __syncthreads();
  }
  #pragma unroll
  for (int j = 0; j < 4; ++j) {
    const int o = o0 + j * 16 + col;
    const float bo = bias[o];
    if (transposed) {                  // kT[bh][n][c], bh = b*8 + o>>5, c = o&31
      const int bh = bb * HEADS_ + (o >> 5), c = o & 31;
      #pragma unroll
      for (int v = 0; v < 8; ++v) {
        int n = p0 + wave * 16 + hlf * 8 + v;
        out[((size_t)bh * NS_ + n) * HC_ + c] = (_Float16)(acc[j][v] + bo);
      }
    } else {                           // vC[(b*C+o)][n], contiguous along n
      _Float16* op = out + ((size_t)bb * C_ + o) * NS_ + p0 + wave * 16 + hlf * 8;
      #pragma unroll
      for (int v = 0; v < 8; ++v) op[v] = (_Float16)(acc[j][v] + bo);
    }
  }
}

// ---------------------------------------------------------------------------
// Offset branch: depthwise 3x3 stride-2 + LayerNorm + GELU + 1x1->2 + tanh
// -> pos (ws + d_out), ref (d_out). One block (64 thr = channels) per (bg,ky,kx).
// ---------------------------------------------------------------------------
__global__ void __launch_bounds__(64) offset_kernel(const float* __restrict__ q,
                                                    const float* __restrict__ w_dw,
                                                    const float* __restrict__ b_dw,
                                                    const float* __restrict__ ln_g,
                                                    const float* __restrict__ ln_b,
                                                    const float* __restrict__ w_off,
                                                    float* __restrict__ pos_ws,
                                                    float* __restrict__ out_pos,
                                                    float* __restrict__ out_ref) {
  __shared__ float sb[GC_];
  const int t   = threadIdx.x;
  const int idx = blockIdx.x;
  const int kx  = idx % WK_;
  const int ky  = (idx / WK_) % HK_;
  const int bg  = idx / (HK_ * WK_);

  const float* qc = q + ((size_t)bg * GC_ + t) * HW_;   // [16][64][48][48] view
  const float* wd = w_dw + t * 9;
  float s = b_dw[t];
  const int iy0 = 2 * ky - 1, ix0 = 2 * kx - 1;
  #pragma unroll
  for (int dy = 0; dy < 3; ++dy) {
    int iy = iy0 + dy;
    if (iy < 0 || iy >= H_) continue;
    #pragma unroll
    for (int dx = 0; dx < 3; ++dx) {
      int ix = ix0 + dx;
      if (ix < 0 || ix >= W_) continue;
      s += qc[iy * W_ + ix] * wd[dy * 3 + dx];
    }
  }
  sb[t] = s;
  __syncthreads();
  float mu = 0.f;
  for (int i = 0; i < GC_; ++i) mu += sb[i];
  mu *= (1.0f / GC_);
  float var = 0.f;
  for (int i = 0; i < GC_; ++i) { float d = sb[i] - mu; var += d * d; }
  var *= (1.0f / GC_);
  float xn = (s - mu) * rsqrtf(var + EPS_) * ln_g[t] + ln_b[t];
  float gl = 0.5f * xn * (1.0f + erff(xn * 0.70710678118654752f));
  __syncthreads();
  sb[t] = gl;
  __syncthreads();
  if (t < 2) {
    float acc = 0.f;
    for (int i = 0; i < GC_; ++i) acc += w_off[t * GC_ + i] * sb[i];
    const float orange = 2.0f / (HK_ - 1.0f);             // FACTOR * 1/(Hk-1)
    float off = tanhf(acc) * orange;
    float refv = (t == 0) ? ((0.5f + ky) / (HK_ - 1.0f) * 2.0f - 1.0f)
                          : ((0.5f + kx) / (WK_ - 1.0f) * 2.0f - 1.0f);
    float p = off + refv;
    size_t o = ((size_t)bg * NS_ + ky * WK_ + kx) * 2 + t;
    pos_ws[o]  = p;
    out_pos[o] = p;
    out_ref[o] = refv;
  }
}

// ---------------------------------------------------------------------------
// grid_sample of x at pos -> x_s [B][C][NS] fp32. One block per (bg, n).
// ---------------------------------------------------------------------------
__global__ void __launch_bounds__(64) sample_kernel(const float* __restrict__ x,
                                                    const float* __restrict__ pos_ws,
                                                    float* __restrict__ x_s) {
  const int t   = threadIdx.x;
  const int idx = blockIdx.x;
  const int n   = idx % NS_;
  const int bg  = idx / NS_;
  const int b   = bg >> 2, g = bg & 3;
  const float py = pos_ws[((size_t)bg * NS_ + n) * 2 + 0];
  const float px = pos_ws[((size_t)bg * NS_ + n) * 2 + 1];
  const float fx = (px + 1.0f) * 0.5f * (W_ - 1);
  const float fy = (py + 1.0f) * 0.5f * (H_ - 1);
  const float x0f = floorf(fx), y0f = floorf(fy);
  const int   x0 = (int)x0f,   y0 = (int)y0f;
  const float wx = fx - x0f,   wy = fy - y0f;
  const float* xc = x + ((size_t)b * C_ + g * GC_ + t) * HW_;
  auto gat = [&](int ix, int iy) -> float {
    if (ix < 0 || ix > W_ - 1 || iy < 0 || iy > H_ - 1) return 0.0f;
    return xc[iy * W_ + ix];
  };
  float v = gat(x0, y0)       * (1 - wx) * (1 - wy)
          + gat(x0 + 1, y0)   * wx       * (1 - wy)
          + gat(x0, y0 + 1)   * (1 - wx) * wy
          + gat(x0 + 1, y0+1) * wx       * wy;
  x_s[((size_t)b * C_ + g * GC_ + t) * NS_ + n] = v;
}

// ---------------------------------------------------------------------------
// Fused attention: S = qK*scale + bias(rpe bilinear), softmax, O = P*V^T.
// Block = 64 thr = 2 waves; each wave owns a 16-row query tile.
// Score tile kept in LDS f16 (never spilled to HBM). Grid (HW/32, B*HEADS).
// ---------------------------------------------------------------------------
__global__ void __launch_bounds__(64) attn_kernel(const float* __restrict__ q,
                                                  const _Float16* __restrict__ kT,
                                                  const _Float16* __restrict__ vC,
                                                  const float* __restrict__ pos_ws,
                                                  const float* __restrict__ rpe,
                                                  float* __restrict__ attn_out) {
  __shared__ _Float16 Pls[2][16][NS_ + 16];
  __shared__ float    rsum[2][16];
  const int tid  = threadIdx.x;
  const int wave = tid >> 5, lane = tid & 31;
  const int col  = lane & 15;
  const int hlf  = lane >> 4;
  const int kb   = hlf * 8;
  const int bh   = blockIdx.y;
  const int b = bh >> 3, h = bh & 7, g = h >> 1, gh = h & 1;
  const int m0 = blockIdx.x * 32 + wave * 16;

  // A frag: q rows (loaded once; K = HC = 32 exactly one WMMA depth)
  const int mrow = m0 + col;
  const float* qb = q + ((size_t)b * C_ + h * HC_) * HW_ + mrow;
  v16h aq;
  #pragma unroll
  for (int i = 0; i < 8; ++i) aq[i]     = (_Float16)qb[(size_t)(kb + i) * HW_];
  #pragma unroll
  for (int i = 0; i < 8; ++i) aq[8 + i] = (_Float16)qb[(size_t)(kb + 16 + i) * HW_];

  const float* rp   = rpe + (size_t)(g * GH_ + gh) * RT_ * RT_;
  const float* posb = pos_ws + (size_t)(b * G_ + g) * NS_ * 2;

  // ---- pass 1: scores + inline rpe bias ----
  for (int nt = 0; nt < NS_ / 16; ++nt) {
    const int n0 = nt * 16;
    const _Float16* kp = kT + ((size_t)bh * NS_ + n0 + col) * HC_;
    if (nt + 1 < NS_ / 16)
      __builtin_prefetch(kT + ((size_t)bh * NS_ + n0 + 16 + col) * HC_, 0, 1);
    v16h bk;
    #pragma unroll
    for (int i = 0; i < 8; ++i) bk[i]     = kp[kb + i];
    #pragma unroll
    for (int i = 0; i < 8; ++i) bk[8 + i] = kp[kb + 16 + i];
    v8f s = {};
    s = __builtin_amdgcn_wmma_f32_16x16x32_f16(false, aq, false, bk,
                                               (short)0, s, false, false);
    const int n_idx = n0 + col;
    const float py = posb[n_idx * 2 + 0];
    const float px = posb[n_idx * 2 + 1];
    #pragma unroll
    for (int v = 0; v < 8; ++v) {
      const int row = v + hlf * 8;
      const int m   = m0 + row;
      const int qy = m / W_, qx = m % W_;
      const float qgy = qy * (2.0f / (H_ - 1.0f)) - 1.0f;
      const float qgx = qx * (2.0f / (W_ - 1.0f)) - 1.0f;
      const float dy = (qgy - py) * 0.5f;
      const float dx = (qgx - px) * 0.5f;
      const float fx = (dx + 1.0f) * 0.5f * (RT_ - 1);
      const float fy = (dy + 1.0f) * 0.5f * (RT_ - 1);
      const float x0f = floorf(fx), y0f = floorf(fy);
      const int   x0 = (int)x0f,   y0 = (int)y0f;
      const float wx = fx - x0f,   wy = fy - y0f;
      const bool vx0 = (x0 >= 0) & (x0 < RT_);
      const bool vx1 = (x0 + 1 >= 0) & (x0 + 1 < RT_);
      const bool vy0 = (y0 >= 0) & (y0 < RT_);
      const bool vy1 = (y0 + 1 >= 0) & (y0 + 1 < RT_);
      const float g00 = (vx0 & vy0) ? rp[y0 * RT_ + x0]           : 0.0f;
      const float g01 = (vx1 & vy0) ? rp[y0 * RT_ + x0 + 1]       : 0.0f;
      const float g10 = (vx0 & vy1) ? rp[(y0 + 1) * RT_ + x0]     : 0.0f;
      const float g11 = (vx1 & vy1) ? rp[(y0 + 1) * RT_ + x0 + 1] : 0.0f;
      const float bias = g00 * (1 - wx) * (1 - wy) + g01 * wx * (1 - wy)
                       + g10 * (1 - wx) * wy       + g11 * wx * wy;
      Pls[wave][row][n_idx] = (_Float16)(s[v] * SCALE_ + bias);
    }
  }
  __syncthreads();

  // ---- pass 2: softmax (2 lanes per row, exp written in place as f16) ----
  {
    const int row = col;
    float mx = -1e30f;
    for (int n = hlf * (NS_ / 2); n < (hlf + 1) * (NS_ / 2); ++n)
      mx = fmaxf(mx, (float)Pls[wave][row][n]);
    mx = fmaxf(mx, __shfl_xor(mx, 16, 32));
    float sum = 0.0f;
    for (int n = hlf * (NS_ / 2); n < (hlf + 1) * (NS_ / 2); ++n) {
      float e = __expf((float)Pls[wave][row][n] - mx);
      sum += e;
      Pls[wave][row][n] = (_Float16)e;
    }
    sum += __shfl_xor(sum, 16, 32);
    if (hlf == 0) rsum[wave][row] = sum;
  }
  __syncthreads();

  // ---- pass 3: O = P * V^T  (two 16-wide channel halves) ----
  v8f o0f = {}, o1f = {};
  for (int k2 = 0; k2 < NS_; k2 += 32) {
    const _Float16* pr  = &Pls[wave][col][k2 + kb];
    const _Float16* vp0 = vC + ((size_t)b * C_ + h * HC_ + col) * NS_ + k2 + kb;
    const _Float16* vp1 = vp0 + (size_t)16 * NS_;
    v16h ap, bv0, bv1;
    #pragma unroll
    for (int i = 0; i < 8; ++i) { ap[i] = pr[i];  bv0[i] = vp0[i];  bv1[i] = vp1[i]; }
    #pragma unroll
    for (int i = 0; i < 8; ++i) {
      ap[8 + i]  = pr[16 + i];
      bv0[8 + i] = vp0[16 + i];
      bv1[8 + i] = vp1[16 + i];
    }
    o0f = __builtin_amdgcn_wmma_f32_16x16x32_f16(false, ap, false, bv0,
                                                 (short)0, o0f, false, false);
    o1f = __builtin_amdgcn_wmma_f32_16x16x32_f16(false, ap, false, bv1,
                                                 (short)0, o1f, false, false);
  }
  #pragma unroll
  for (int v = 0; v < 8; ++v) {
    const int row = v + hlf * 8;
    const float inv = 1.0f / rsum[wave][row];
    const int m = m0 + row;
    attn_out[((size_t)b * C_ + h * HC_ + col) * HW_ + m]      = o0f[v] * inv;
    attn_out[((size_t)b * C_ + h * HC_ + 16 + col) * HW_ + m] = o1f[v] * inv;
  }
}

// ---------------------------------------------------------------------------
extern "C" void kernel_launch(void* const* d_in, const int* in_sizes, int n_in,
                              void* d_out, int out_size, void* d_ws, size_t ws_size,
                              hipStream_t stream) {
  (void)in_sizes; (void)n_in; (void)out_size; (void)ws_size;
  const float* x    = (const float*)d_in[0];
  const float* wq   = (const float*)d_in[1];
  const float* bq   = (const float*)d_in[2];
  const float* w_dw = (const float*)d_in[3];
  const float* b_dw = (const float*)d_in[4];
  const float* ln_g = (const float*)d_in[5];
  const float* ln_b = (const float*)d_in[6];
  const float* woff = (const float*)d_in[7];
  const float* wk   = (const float*)d_in[8];
  const float* bk   = (const float*)d_in[9];
  const float* wv   = (const float*)d_in[10];
  const float* bv   = (const float*)d_in[11];
  const float* wo   = (const float*)d_in[12];
  const float* bo   = (const float*)d_in[13];
  const float* rpe  = (const float*)d_in[14];

  float* y       = (float*)d_out;
  float* out_pos = y + (size_t)B_ * C_ * HW_;                 // [B,G,Hk,Wk,2]
  float* out_ref = out_pos + (size_t)B_ * G_ * NS_ * 2;

  char* ws = (char*)d_ws;
  float*    q_ws   = (float*)(ws + 0);          //  9.44 MB  [B,C,HW]
  float*    ao_ws  = (float*)(ws + 9437184);    //  9.44 MB  [B,C,HW]
  float*    xs_ws  = (float*)(ws + 18874368);   //  2.36 MB  [B,C,NS]
  float*    pos_ws = (float*)(ws + 21233664);   //  73.7 KB  [BG,NS,2]
  _Float16* kT     = (_Float16*)(ws + 21307392);//  1.18 MB  [BH,NS,HC]
  _Float16* vC     = (_Float16*)(ws + 22487040);//  1.18 MB  [B,C,NS]

  // 1) q projection
  proj_kernel<HW_><<<dim3(HW_ / 64, C_ / 64, B_), 128, 0, stream>>>(x, wq, bq, q_ws);
  // 2) offset branch -> pos (+ pos/ref outputs)
  offset_kernel<<<dim3(B_ * G_ * HK_ * WK_), 64, 0, stream>>>(
      q_ws, w_dw, b_dw, ln_g, ln_b, woff, pos_ws, out_pos, out_ref);
  // 3) sample x at deformed points
  sample_kernel<<<dim3(B_ * G_ * NS_), 64, 0, stream>>>(x, pos_ws, xs_ws);
  // 4) k / v projections (f16, WMMA-friendly layouts)
  kv_kernel<<<dim3(NS_ / 64, C_ / 64, B_), 128, 0, stream>>>(xs_ws, wk, bk, kT, 1);
  kv_kernel<<<dim3(NS_ / 64, C_ / 64, B_), 128, 0, stream>>>(xs_ws, wv, bv, vC, 0);
  // 5) fused attention (QK + rpe-bias grid_sample + softmax + PV)
  attn_kernel<<<dim3(HW_ / 32, B_ * HEADS_), 64, 0, stream>>>(
      q_ws, kT, vC, pos_ws, rpe, ao_ws);
  // 6) output projection
  proj_kernel<HW_><<<dim3(HW_ / 64, C_ / 64, B_), 128, 0, stream>>>(ao_ws, wo, bo, y);
}
